// Graph_Attention_Union3_76879914598916
// MI455X (gfx1250) — matrix operand order repaired and checked by reading, not compiled, not run on hardware
//
#include <hip/hip_runtime.h>
#include <hip/hip_bf16.h>

typedef __bf16 bf16;
typedef __attribute__((ext_vector_type(16))) __bf16 v16bf;
typedef __attribute__((ext_vector_type(8)))  float  v8f;

#define NCH   256     // C
#define NXX   625     // Hx*Wx
#define NZZ   169     // Hz*Wz
#define NZP   192     // Nz padded to multiple of 32 (WMMA K-step), zero-filled
#define NB    128     // batch

union ABFrag { uint4 u[2]; v16bf v; };
union Pack8  { uint4 u; bf16 h[8]; };

__device__ __forceinline__ v8f wmma_bf16(v16bf a, v16bf b, v8f c) {
  // D = A(16x32) * B(32x16) + C(16x16 f32)
  return __builtin_amdgcn_wmma_f32_16x16x32_bf16(false, a, false, b, (short)0, c,
                                                 false, false);
}

// ---------------------------------------------------------------------------
// Prep: cast weights to bf16, fold conv-bias + BatchNorm into (scale, bias).
// ---------------------------------------------------------------------------
__global__ void __launch_bounds__(256)
prep_kernel(const float* __restrict__ Wq, const float* __restrict__ Ws,
            const float* __restrict__ Wg, const float* __restrict__ Wf,
            const float* __restrict__ bg, const float* __restrict__ bfin,
            const float* __restrict__ ggamma, const float* __restrict__ gbeta,
            const float* __restrict__ gmean,  const float* __restrict__ gvar,
            const float* __restrict__ fgamma, const float* __restrict__ fbeta,
            const float* __restrict__ fmean,  const float* __restrict__ fvar,
            bf16* __restrict__ Wq_bf, bf16* __restrict__ Ws_bf,
            bf16* __restrict__ Wg_bf, bf16* __restrict__ Wf_bf,
            float* __restrict__ gscale, float* __restrict__ gbias,
            float* __restrict__ fscale, float* __restrict__ fbias)
{
  const int tid = blockIdx.x * blockDim.x + threadIdx.x;
  const int nth = gridDim.x * blockDim.x;
  for (int i = tid; i < NCH * NCH; i += nth) {
    Wq_bf[i] = (bf16)Wq[i];
    Ws_bf[i] = (bf16)Ws[i];
    Wg_bf[i] = (bf16)Wg[i];
  }
  for (int i = tid; i < NCH * 2 * NCH; i += nth) Wf_bf[i] = (bf16)Wf[i];
  for (int o = tid; o < NCH; o += nth) {
    float gi = ggamma[o] * rsqrtf(gvar[o] + 1e-5f);
    gscale[o] = gi;
    gbias[o]  = bg[o] * gi + gbeta[o] - gmean[o] * gi;
    float fi = fgamma[o] * rsqrtf(fvar[o] + 1e-5f);
    fscale[o] = fi;
    fbias[o]  = bfin[o] * fi + fbeta[o] - fmean[o] * fi;
  }
}

// ---------------------------------------------------------------------------
// Fused dual conv1x1: from X[b][c][n] (f32) produce
//   outq_t[b][n][o] = sum_c Wq[o][c]*X[c][n] + bq[o]            (transposed!)
//   outg   =          relu(gs[o]*(sum_c Wg[o][c]*X[c][n]) + gb[o])
//     TG=1: [b][n][o] transposed (b128 stores); TG=0: [b][o][GSTRIDE] strided.
// One wave: 4 output-channel tiles x 1 pixel tile; B fragment shared by all
// 8 WMMAs per K-step. N compile-time => strided loads get immediate offsets.
// ---------------------------------------------------------------------------
template <int N, bool TG, int GSTRIDE>
__global__ void __launch_bounds__(256)
wmma_conv_dual(const float* __restrict__ X,
               const bf16* __restrict__ Wq, const bf16* __restrict__ Wg,
               const float* __restrict__ bq,
               const float* __restrict__ gs, const float* __restrict__ gb,
               bf16* __restrict__ outq_t, bf16* __restrict__ outg)
{
  const int lane = threadIdx.x & 31;
  const int wave = threadIdx.x >> 5;
  const int og   = blockIdx.x * 64;                  // 4 o-tiles
  const int nb   = (blockIdx.y * 8 + wave) * 16;     // pixel tile
  if (nb >= N) return;                                // wave-uniform
  const int b = blockIdx.z;
  const float* Xb = X + (size_t)b * NCH * N;

  const int m   = lane & 15;
  const int kbA = (lane < 16) ? 0 : 8;
  const int kbB = (lane < 16) ? 0 : 8;
  const int n   = nb + m;
  const bool nv = (n < N);
  const int nc  = nv ? n : (N - 1);

  v8f aq[4] = {};
  v8f ag[4] = {};
  for (int k0 = 0; k0 < NCH; k0 += 32) {
    v16bf Bv;
    const float* xp = Xb + (size_t)(k0 + kbB) * N + nc;
    __builtin_prefetch(xp + 32 * N, 0, 1);
#pragma unroll
    for (int e = 0; e < 16; ++e) Bv[e] = (bf16)xp[(e + (e & 8)) * N];  // imm offsets
#pragma unroll
    for (int t = 0; t < 4; ++t) {
      ABFrag A;
      const bf16* wr = Wq + (size_t)(og + t * 16 + m) * NCH + k0 + kbA;
      A.u[0] = *reinterpret_cast<const uint4*>(wr);
      A.u[1] = *reinterpret_cast<const uint4*>(wr + 16);
      aq[t] = wmma_bf16(A.v, Bv, aq[t]);
      const bf16* gr = Wg + (size_t)(og + t * 16 + m) * NCH + k0 + kbA;
      A.u[0] = *reinterpret_cast<const uint4*>(gr);
      A.u[1] = *reinterpret_cast<const uint4*>(gr + 16);
      ag[t] = wmma_bf16(A.v, Bv, ag[t]);
    }
  }
  if (!nv) return;
  const int r0 = (lane < 16) ? 0 : 8;

  bf16* qrow = outq_t + ((size_t)b * N + n) * NCH + og;
#pragma unroll
  for (int t = 0; t < 4; ++t) {
    Pack8 st;
#pragma unroll
    for (int r = 0; r < 8; ++r) st.h[r] = (bf16)(aq[t][r] + bq[og + t * 16 + r0 + r]);
    *reinterpret_cast<uint4*>(qrow + t * 16 + r0) = st.u;   // 8 contiguous channels
  }
  if constexpr (TG) {
    bf16* grow = outg + ((size_t)b * N + n) * NCH + og;
#pragma unroll
    for (int t = 0; t < 4; ++t) {
      Pack8 st;
#pragma unroll
      for (int r = 0; r < 8; ++r) {
        const int o = og + t * 16 + r0 + r;
        st.h[r] = (bf16)fmaxf(ag[t][r] * gs[o] + gb[o], 0.0f);
      }
      *reinterpret_cast<uint4*>(grow + t * 16 + r0) = st.u;
    }
  } else {
#pragma unroll
    for (int t = 0; t < 4; ++t)
#pragma unroll
      for (int r = 0; r < 8; ++r) {
        const int o = og + t * 16 + r0 + r;
        outg[((size_t)b * NCH + o) * GSTRIDE + n] =
            (bf16)fmaxf(ag[t][r] * gs[o] + gb[o], 0.0f);
      }
  }
}

// ---------------------------------------------------------------------------
// similar[b][n][m] = sum_c xq_t[b][n][c] * zs_t[b][m][c]
// Both operands K-major => every fragment is two aligned uint4 loads.
// ---------------------------------------------------------------------------
__global__ void __launch_bounds__(256)
wmma_similar(const bf16* __restrict__ xq_t, const bf16* __restrict__ zs_t,
             float* __restrict__ sim)
{
  const int lane = threadIdx.x & 31;
  const int wave = threadIdx.x >> 5;
  const int nxb  = blockIdx.x * 16;
  const int mzb  = (blockIdx.y * 8 + wave) * 16;
  if (mzb >= NZZ) return;
  const int b = blockIdx.z;

  const int m   = lane & 15;
  const int kbA = (lane < 16) ? 0 : 8;
  const int kbB = (lane < 16) ? 0 : 8;
  const int nx  = nxb + m;  const int nxc = (nx < NXX) ? nx : NXX - 1;
  const int mz  = mzb + m;  const bool mzv = (mz < NZZ);
  const int mzc = mzv ? mz : NZZ - 1;

  const bf16* xr = xq_t + ((size_t)b * NXX + nxc) * NCH;
  const bf16* zr = zs_t + ((size_t)b * NZZ + mzc) * NCH;

  v8f acc = {};
  for (int k0 = 0; k0 < NCH; k0 += 32) {
    ABFrag A, Bv;
    A.u[0]  = *reinterpret_cast<const uint4*>(xr + k0 + kbA);
    A.u[1]  = *reinterpret_cast<const uint4*>(xr + k0 + kbA + 16);
    Bv.u[0] = *reinterpret_cast<const uint4*>(zr + k0 + kbB);
    Bv.u[1] = *reinterpret_cast<const uint4*>(zr + k0 + kbB + 16);
    acc = wmma_bf16(A.v, Bv.v, acc);
  }
  if (mzv) {
    const int r0 = (lane < 16) ? 0 : 8;
#pragma unroll
    for (int r = 0; r < 8; ++r) {
      const int nrow = nxb + r0 + r;
      if (nrow < NXX)
        sim[((size_t)b * NXX + nrow) * NZZ + mz] = acc[r];
    }
  }
}

// ---------------------------------------------------------------------------
// Full sparsemax over Nz=169 (TOPK=256 >= Nz => degenerates to sparsemax).
// tau solved by bisection on f(tau)=sum max(x-tau,0)=1, tau in [max-1,max].
// One wave per row; emits bf16 score padded to 192 with zeros.
// ---------------------------------------------------------------------------
__global__ void __launch_bounds__(256)
sparsemax_kernel(const float* __restrict__ sim, bf16* __restrict__ score)
{
  const int lane = threadIdx.x & 31;
  const int row  = (blockIdx.x * blockDim.x + threadIdx.x) >> 5;  // b*Nx + n
  if (row >= NB * NXX) return;
  const float* x = sim + (size_t)row * NZZ;

  float v[6];
  float mx = -1e30f;
#pragma unroll
  for (int i = 0; i < 6; ++i) {
    const int mi = lane + 32 * i;
    v[i] = (mi < NZZ) ? x[mi] : -1e30f;
    mx = fmaxf(mx, v[i]);
  }
#pragma unroll
  for (int off = 16; off; off >>= 1) mx = fmaxf(mx, __shfl_xor(mx, off, 32));

  float lo = mx - 1.0f, hi = mx;
  for (int it = 0; it < 26; ++it) {
    const float tau = 0.5f * (lo + hi);
    float s = 0.0f;
#pragma unroll
    for (int i = 0; i < 6; ++i) s += fmaxf(v[i] - tau, 0.0f);
#pragma unroll
    for (int off = 16; off; off >>= 1) s += __shfl_xor(s, off, 32);
    if (s >= 1.0f) lo = tau; else hi = tau;
  }
  const float tau = 0.5f * (lo + hi);

  bf16* o = score + (size_t)row * NZP;
#pragma unroll
  for (int i = 0; i < 6; ++i)                       // 6*32 == NZP exactly
    o[lane + 32 * i] = (bf16)fmaxf(v[i] - tau, 0.0f);
}

// ---------------------------------------------------------------------------
// embedding_t[b][n][c] = sum_m zg[b][c][m] * score[b][n][m], K padded to 192.
// A = zg [C][NZP] contiguous-K, B = score [Nx][NZP] contiguous-K; 4 c-tiles
// per wave share one B fragment. Output transposed with b128 stores.
// ---------------------------------------------------------------------------
__global__ void __launch_bounds__(256)
wmma_embed(const bf16* __restrict__ zg, const bf16* __restrict__ sc,
           bf16* __restrict__ emb_t)
{
  const int lane = threadIdx.x & 31;
  const int wave = threadIdx.x >> 5;
  const int cg   = blockIdx.x * 64;                  // 4 c-tiles
  const int nb   = (blockIdx.y * 8 + wave) * 16;
  if (nb >= NXX) return;
  const int b = blockIdx.z;
  const bf16* zgb = zg + (size_t)b * NCH * NZP;

  const int m   = lane & 15;
  const int kbA = (lane < 16) ? 0 : 8;
  const int kbB = (lane < 16) ? 0 : 8;
  const int n   = nb + m;
  const bool nv = (n < NXX);
  const int nc  = nv ? n : (NXX - 1);
  const bf16* sr = sc + ((size_t)b * NXX + nc) * NZP;

  v8f acc[4] = {};
  for (int k0 = 0; k0 < NZP; k0 += 32) {
    ABFrag Bv;
    Bv.u[0] = *reinterpret_cast<const uint4*>(sr + k0 + kbB);
    Bv.u[1] = *reinterpret_cast<const uint4*>(sr + k0 + kbB + 16);
#pragma unroll
    for (int t = 0; t < 4; ++t) {
      ABFrag A;
      const bf16* ar = zgb + (size_t)(cg + t * 16 + m) * NZP + k0 + kbA;
      A.u[0] = *reinterpret_cast<const uint4*>(ar);
      A.u[1] = *reinterpret_cast<const uint4*>(ar + 16);
      acc[t] = wmma_bf16(A.v, Bv.v, acc[t]);
    }
  }
  if (!nv) return;
  const int r0 = (lane < 16) ? 0 : 8;
  bf16* erow = emb_t + ((size_t)b * NXX + n) * NCH + cg;
#pragma unroll
  for (int t = 0; t < 4; ++t) {
    Pack8 st;
#pragma unroll
    for (int r = 0; r < 8; ++r) st.h[r] = (bf16)acc[t][r];
    *reinterpret_cast<uint4*>(erow + t * 16 + r0) = st.u;
  }
}

// ---------------------------------------------------------------------------
// Final: out[b][o][n] = relu(fs[o]*(sum_{k<512} Wf[o][k]*cat[k][n]) + fb[o]),
// cat = [embedding ; xf_g], both K-major transposed => contiguous B loads.
// 4 o-tiles per wave share one B fragment. f32 output to d_out.
// ---------------------------------------------------------------------------
__global__ void __launch_bounds__(256)
wmma_final(const bf16* __restrict__ emb_t, const bf16* __restrict__ xg_t,
           const bf16* __restrict__ Wf, const float* __restrict__ fs,
           const float* __restrict__ fb, float* __restrict__ out)
{
  const int lane = threadIdx.x & 31;
  const int wave = threadIdx.x >> 5;
  const int og   = blockIdx.x * 64;                  // 4 o-tiles
  const int nb   = (blockIdx.y * 8 + wave) * 16;
  if (nb >= NXX) return;
  const int b = blockIdx.z;

  const int m   = lane & 15;
  const int kbA = (lane < 16) ? 0 : 8;
  const int kbB = (lane < 16) ? 0 : 8;
  const int n   = nb + m;
  const bool nv = (n < NXX);
  const int nc  = nv ? n : (NXX - 1);
  const bf16* er = emb_t + ((size_t)b * NXX + nc) * NCH;
  const bf16* xr = xg_t  + ((size_t)b * NXX + nc) * NCH;

  v8f acc[4] = {};
  for (int k0 = 0; k0 < 2 * NCH; k0 += 32) {
    const bf16* br = ((k0 < NCH) ? (er + k0) : (xr + (k0 - NCH))) + kbB;
    ABFrag Bv;
    Bv.u[0] = *reinterpret_cast<const uint4*>(br);
    Bv.u[1] = *reinterpret_cast<const uint4*>(br + 16);
#pragma unroll
    for (int t = 0; t < 4; ++t) {
      ABFrag A;
      const bf16* wr = Wf + (size_t)(og + t * 16 + m) * (2 * NCH) + k0 + kbA;
      A.u[0] = *reinterpret_cast<const uint4*>(wr);
      A.u[1] = *reinterpret_cast<const uint4*>(wr + 16);
      acc[t] = wmma_bf16(A.v, Bv.v, acc[t]);
    }
  }
  if (!nv) return;
  const int r0 = (lane < 16) ? 0 : 8;
#pragma unroll
  for (int t = 0; t < 4; ++t)
#pragma unroll
    for (int r = 0; r < 8; ++r) {
      const int o = og + t * 16 + r0 + r;
      out[((size_t)b * NCH + o) * NXX + n] = fmaxf(acc[t][r] * fs[o] + fb[o], 0.0f);
    }
}

// ---------------------------------------------------------------------------
extern "C" void kernel_launch(void* const* d_in, const int* in_sizes, int n_in,
                              void* d_out, int out_size, void* d_ws, size_t ws_size,
                              hipStream_t stream) {
  (void)in_sizes; (void)n_in; (void)out_size; (void)ws_size;
  const float* zf   = (const float*)d_in[0];
  const float* xf   = (const float*)d_in[1];
  const float* Wq   = (const float*)d_in[2];
  const float* bq   = (const float*)d_in[3];
  const float* Ws   = (const float*)d_in[4];
  const float* bs   = (const float*)d_in[5];
  const float* Wg   = (const float*)d_in[6];
  const float* bg   = (const float*)d_in[7];
  const float* g_ga = (const float*)d_in[8];
  const float* g_be = (const float*)d_in[9];
  const float* g_me = (const float*)d_in[10];
  const float* g_va = (const float*)d_in[11];
  const float* Wf   = (const float*)d_in[12];
  const float* bfin = (const float*)d_in[13];
  const float* f_ga = (const float*)d_in[14];
  const float* f_be = (const float*)d_in[15];
  const float* f_me = (const float*)d_in[16];
  const float* f_va = (const float*)d_in[17];
  float* out = (float*)d_out;

  char* ws = (char*)d_ws;
  size_t off = 0;
  auto take = [&](size_t bytes) -> void* {
    void* p = ws + off;
    off += (bytes + 255) & ~(size_t)255;
    return p;
  };

  bf16*  Wq_bf = (bf16*)take((size_t)NCH * NCH * 2);
  bf16*  Ws_bf = (bf16*)take((size_t)NCH * NCH * 2);
  bf16*  Wg_bf = (bf16*)take((size_t)NCH * NCH * 2);
  bf16*  Wf_bf = (bf16*)take((size_t)NCH * 2 * NCH * 2);
  float* gscale = (float*)take(NCH * 4);
  float* gbias  = (float*)take(NCH * 4);
  float* fscale = (float*)take(NCH * 4);
  float* fbias  = (float*)take(NCH * 4);
  bf16*  xq_t = (bf16*)take((size_t)NB * NXX * NCH * 2);  // xf_trans, [b][n][c]
  bf16*  xg_t = (bf16*)take((size_t)NB * NXX * NCH * 2);  // xf_g,     [b][n][c]
  bf16*  zs_t = (bf16*)take((size_t)NB * NZZ * NCH * 2);  // zf_trans, [b][m][c]
  bf16*  zg   = (bf16*)take((size_t)NB * NCH * NZP * 2);  // zf_g,     [b][c][m] K-pad
  float* sim  = (float*)take((size_t)NB * NXX * NZZ * 4); // similarity logits
  bf16*  sc   = (bf16*)take((size_t)NB * NXX * NZP * 2);  // sparsemax score
  bf16*  emb_t = (bf16*)take((size_t)NB * NXX * NCH * 2); // embedding, [b][n][c]

  // zero the K-pad of zf_g once (conv writes only m < 169)
  hipMemsetAsync(zg, 0, (size_t)NB * NCH * NZP * 2, stream);

  prep_kernel<<<64, 256, 0, stream>>>(Wq, Ws, Wg, Wf, bg, bfin,
                                      g_ga, g_be, g_me, g_va,
                                      f_ga, f_be, f_me, f_va,
                                      Wq_bf, Ws_bf, Wg_bf, Wf_bf,
                                      gscale, gbias, fscale, fbias);

  // fused dual-weight conv passes (Wq/Ws share B fragments with Wg)
  wmma_conv_dual<NXX, true, 0><<<dim3(4, 5, NB), 256, 0, stream>>>(
      xf, Wq_bf, Wg_bf, bq, gscale, gbias, xq_t, xg_t);
  wmma_conv_dual<NZZ, false, NZP><<<dim3(4, 2, NB), 256, 0, stream>>>(
      zf, Ws_bf, Wg_bf, bs, gscale, gbias, zs_t, zg);

  // similarity logits: [B, 625, 169], K = 256
  wmma_similar<<<dim3(40, 2, NB), 256, 0, stream>>>(xq_t, zs_t, sim);

  // sparse attention: one wave per (b, n) row
  sparsemax_kernel<<<(NB * NXX) / 8, 256, 0, stream>>>(sim, sc);

  // embedding: [B, 625, 256] transposed, K = 192 (padded)
  wmma_embed<<<dim3(4, 5, NB), 256, 0, stream>>>(zg, sc, emb_t);

  // fused concat + final conv + BN + ReLU, K = 512, f32 output
  wmma_final<<<dim3(4, 5, NB), 256, 0, stream>>>(emb_t, xg_t, Wf_bf, fscale,
                                                 fbias, out);
}
